// ResidualGCNLayer_90958817394877
// MI455X (gfx1250) — matrix operand order repaired and checked by reading, not compiled, not run on hardware
//
#include <hip/hip_runtime.h>

// ---------------------------------------------------------------------------
// Types for gfx1250 WMMA
// ---------------------------------------------------------------------------
typedef __attribute__((ext_vector_type(16))) __bf16 v16bf;
typedef __attribute__((ext_vector_type(8)))  __bf16 bf16x8;
typedef __attribute__((ext_vector_type(8)))  float  v8f;
typedef __attribute__((ext_vector_type(4)))  float  f4;

// pack {hi16(u1), hi16(u0)} into one dword with a single v_perm_b32
static __device__ __forceinline__ unsigned pack_hi16(unsigned u0, unsigned u1) {
  return __builtin_amdgcn_perm(u1, u0, 0x07060302u);
}
static __device__ __forceinline__ unsigned fbits(float f) {
  return __builtin_bit_cast(unsigned, f);
}
static __device__ __forceinline__ float ffloat(unsigned u) {
  return __builtin_bit_cast(float, u);
}

// ---------------------------------------------------------------------------
// Elementwise helpers
// ---------------------------------------------------------------------------
__global__ void k_zero(float* __restrict__ p, int n) {
  int i = blockIdx.x * blockDim.x + threadIdx.x;
  if (i < n) p[i] = 0.0f;
}

__global__ void k_negcopy(const float* __restrict__ x, float* __restrict__ y, int n4) {
  int i = blockIdx.x * blockDim.x + threadIdx.x;
  if (i < n4) {
    f4 v = ((const f4*)x)[i];
    ((f4*)y)[i] = -v;
  }
}

// deg[src[e]] += 1
__global__ void k_deg(const int* __restrict__ src, float* __restrict__ deg, int E) {
  int e = blockIdx.x * blockDim.x + threadIdx.x;
  if (e < E) atomicAdd(&deg[src[e]], 1.0f);
}

// norm[e] = -(dinv[src]*dinv[dst]),  dinv = deg>0 ? rsqrt(max(deg,1)) : 0
__global__ void k_norm(const int* __restrict__ src, const int* __restrict__ dst,
                       const float* __restrict__ deg, float* __restrict__ norm, int E) {
  int e = blockIdx.x * blockDim.x + threadIdx.x;
  if (e < E) {
    float ds = deg[src[e]], dd = deg[dst[e]];
    float a = ds > 0.0f ? rsqrtf(fmaxf(ds, 1.0f)) : 0.0f;
    float b = dd > 0.0f ? rsqrtf(fmaxf(dd, 1.0f)) : 0.0f;
    norm[e] = -(a * b);
  }
}

// out[dst[e], :] += factor * norm[e] * h[src[e], :]   (one block per edge, C=512)
__global__ __launch_bounds__(128) void k_spmm(const int* __restrict__ src,
                                              const int* __restrict__ dst,
                                              const float* __restrict__ norm,
                                              const float* __restrict__ h,
                                              float* __restrict__ out,
                                              float factor, int C) {
  int e = blockIdx.x;
  float w = norm[e] * factor;
  int s = src[e], d = dst[e];
  int c = threadIdx.x << 2;
  f4 v = *(const f4*)(h + (size_t)s * C + c);
  float* o = out + (size_t)d * C + c;
  atomicAdd(o + 0, w * v.x);
  atomicAdd(o + 1, w * v.y);
  atomicAdd(o + 2, w * v.z);
  atomicAdd(o + 3, w * v.w);
}

// ---------------------------------------------------------------------------
// Split-bf16 WMMA GEMM:  C = sum_p A_p @ B_p + bias ; optional BN partial sums
// Block tile 128x64, 8 waves (4x2), each wave 2x2 of 16x16; K stepped by 32.
// hi = trunc16(f), lo = trunc16(f - hi): 3 WMMAs/product gives ~fp32 accuracy.
// Double-buffered LDS, one barrier per k-step; packed b32 LDS stores.
// ---------------------------------------------------------------------------
#define BM 128
#define BN 64
#define BK 32
#define APITCH 40   // halves; 80B row pitch, keeps 16B alignment, staggers banks
#define BPITCH 40

__global__ __launch_bounds__(256) void k_gemm(
    const float* __restrict__ A0, const float* __restrict__ A1, const float* __restrict__ A2,
    const float* __restrict__ B0, const float* __restrict__ B1, const float* __restrict__ B2,
    const float* __restrict__ bias, float* __restrict__ C,
    int M, int Kd, int Ncol, int npairs,
    float* __restrict__ bn_sum, float* __restrict__ bn_sumsq, int do_bn) {
  __shared__ alignas(16) __bf16 AsH[2][BM * APITCH];
  __shared__ alignas(16) __bf16 AsL[2][BM * APITCH];
  __shared__ alignas(16) __bf16 BtH[2][BN * BPITCH];   // transposed: [n][k]
  __shared__ alignas(16) __bf16 BtL[2][BN * BPITCH];

  const int tid  = threadIdx.x;
  const int lane = tid & 31;
  const int wave = tid >> 5;
  const int wm   = wave >> 1;   // 0..3 -> 32-row strip
  const int wn   = wave & 1;    // 0..1 -> 32-col strip
  const int lm   = lane & 15;
  const int hf   = lane >> 4;

  const int m0 = blockIdx.x * BM;
  const int n0 = blockIdx.y * BN;

  // stage one (pair, k0) tile into LDS buffer `buf`
  auto stage = [&](int buf, const float* __restrict__ Ap, const float* __restrict__ Bp,
                   int k0) {
    // ---- A tile: 128x32 f32, 1024 float4 chunks, pack k-pairs into b32 ----
#pragma unroll
    for (int it = 0; it < 4; ++it) {
      int c   = tid + it * 256;
      int row = c >> 3;
      int k   = (c & 7) << 2;
      int gm  = m0 + row;
      f4 v = (f4){0.f, 0.f, 0.f, 0.f};
      if (gm < M) v = *(const f4*)(Ap + (size_t)gm * Kd + k0 + k);
#pragma unroll
      for (int j = 0; j < 2; ++j) {            // two k-pairs
        unsigned u0 = fbits(v[2 * j]);
        unsigned u1 = fbits(v[2 * j + 1]);
        float    l0 = v[2 * j]     - ffloat(u0 & 0xffff0000u);
        float    l1 = v[2 * j + 1] - ffloat(u1 & 0xffff0000u);
        *(unsigned*)&AsH[buf][row * APITCH + k + 2 * j] = pack_hi16(u0, u1);
        *(unsigned*)&AsL[buf][row * APITCH + k + 2 * j] = pack_hi16(fbits(l0), fbits(l1));
      }
    }
    // ---- B tile: 32x64 f32 transposed; thread owns 4 n x 2 k ----
    {
      int k = (tid >> 4) << 1;   // 0,2,..,30
      int n = (tid & 15) << 2;   // 0,4,..,60
      f4 va = *(const f4*)(Bp + (size_t)(k0 + k) * Ncol + n0 + n);
      f4 vb = *(const f4*)(Bp + (size_t)(k0 + k + 1) * Ncol + n0 + n);
#pragma unroll
      for (int j = 0; j < 4; ++j) {
        unsigned u0 = fbits(va[j]);
        unsigned u1 = fbits(vb[j]);
        float    l0 = va[j] - ffloat(u0 & 0xffff0000u);
        float    l1 = vb[j] - ffloat(u1 & 0xffff0000u);
        *(unsigned*)&BtH[buf][(n + j) * BPITCH + k] = pack_hi16(u0, u1);
        *(unsigned*)&BtL[buf][(n + j) * BPITCH + k] = pack_hi16(fbits(l0), fbits(l1));
      }
    }
  };

  v8f acc[2][2];
#pragma unroll
  for (int a = 0; a < 2; ++a)
#pragma unroll
    for (int b = 0; b < 2; ++b) acc[a][b] = (v8f){0.f,0.f,0.f,0.f,0.f,0.f,0.f,0.f};

  const int ksteps = Kd / BK;
  const int total  = npairs * ksteps;

  stage(0, A0, B0, 0);
  __syncthreads();

  int cur = 0;
  for (int it = 0; it < total; ++it) {
    // ---- load fragments from buffer `cur` (ISA per-lane layout) ----
    v16bf aH[2], aL[2], bH[2], bL[2];
#pragma unroll
    for (int mi = 0; mi < 2; ++mi) {
      int m = wm * 32 + mi * 16 + lm;
      bf16x8 h0 = *(const bf16x8*)&AsH[cur][m * APITCH + hf * 8];
      bf16x8 h1 = *(const bf16x8*)&AsH[cur][m * APITCH + 16 + hf * 8];
      aH[mi] = __builtin_shufflevector(h0, h1, 0,1,2,3,4,5,6,7,8,9,10,11,12,13,14,15);
      bf16x8 l0 = *(const bf16x8*)&AsL[cur][m * APITCH + hf * 8];
      bf16x8 l1 = *(const bf16x8*)&AsL[cur][m * APITCH + 16 + hf * 8];
      aL[mi] = __builtin_shufflevector(l0, l1, 0,1,2,3,4,5,6,7,8,9,10,11,12,13,14,15);
    }
#pragma unroll
    for (int ni = 0; ni < 2; ++ni) {
      int n = wn * 32 + ni * 16 + lm;
      bf16x8 h0 = *(const bf16x8*)&BtH[cur][n * BPITCH + hf * 8];
      bf16x8 h1 = *(const bf16x8*)&BtH[cur][n * BPITCH + 16 + hf * 8];
      bH[ni] = __builtin_shufflevector(h0, h1, 0,1,2,3,4,5,6,7,8,9,10,11,12,13,14,15);
      bf16x8 l0 = *(const bf16x8*)&BtL[cur][n * BPITCH + hf * 8];
      bf16x8 l1 = *(const bf16x8*)&BtL[cur][n * BPITCH + 16 + hf * 8];
      bL[ni] = __builtin_shufflevector(l0, l1, 0,1,2,3,4,5,6,7,8,9,10,11,12,13,14,15);
    }

    // ---- stage next tile into the other buffer (overlaps with WMMA) ----
    int nxt = it + 1;
    if (nxt < total) {
      int p  = nxt / ksteps;
      int k0 = (nxt % ksteps) * BK;
      const float* Ap = (p == 0) ? A0 : ((p == 1) ? A1 : A2);
      const float* Bp = (p == 0) ? B0 : ((p == 1) ? B1 : B2);
      stage(cur ^ 1, Ap, Bp, k0);
    }

    // ---- 3-term split-bf16 product: hi*hi + hi*lo + lo*hi ----
#pragma unroll
    for (int mi = 0; mi < 2; ++mi)
#pragma unroll
      for (int ni = 0; ni < 2; ++ni) {
        acc[mi][ni] = __builtin_amdgcn_wmma_f32_16x16x32_bf16(
            false, aH[mi], false, bH[ni], (short)0, acc[mi][ni], false, false);
        acc[mi][ni] = __builtin_amdgcn_wmma_f32_16x16x32_bf16(
            false, aH[mi], false, bL[ni], (short)0, acc[mi][ni], false, false);
        acc[mi][ni] = __builtin_amdgcn_wmma_f32_16x16x32_bf16(
            false, aL[mi], false, bH[ni], (short)0, acc[mi][ni], false, false);
      }

    __syncthreads();
    cur ^= 1;
  }

  // ---- epilogue: bias add, store, BN partial sums ----
#pragma unroll
  for (int mi = 0; mi < 2; ++mi)
#pragma unroll
    for (int ni = 0; ni < 2; ++ni) {
      int gn  = n0 + wn * 32 + ni * 16 + lm;
      int gmb = m0 + wm * 32 + mi * 16 + hf * 8;
      float bv = bias[gn];
      float s1 = 0.f, s2 = 0.f;
#pragma unroll
      for (int v = 0; v < 8; ++v) {
        int gm = gmb + v;
        if (gm < M) {
          float val = acc[mi][ni][v] + bv;
          C[(size_t)gm * Ncol + gn] = val;
          s1 += val;
          s2 += val * val;
        }
      }
      if (do_bn) {
        atomicAdd(&bn_sum[gn], s1);
        atomicAdd(&bn_sumsq[gn], s2);
      }
    }
}

// mean/var -> per-channel affine (scale, shift)
__global__ void k_bnfin(const float* __restrict__ sum, const float* __restrict__ sumsq,
                        const float* __restrict__ gamma, const float* __restrict__ beta,
                        float* __restrict__ scale, float* __restrict__ shift,
                        int M, int C, float eps) {
  int c = blockIdx.x * blockDim.x + threadIdx.x;
  if (c < C) {
    float inv  = 1.0f / (float)M;
    float mean = sum[c] * inv;
    float var  = fmaxf(sumsq[c] * inv - mean * mean, 0.0f);
    float g    = gamma[c] * rsqrtf(var + eps);
    scale[c] = g;
    shift[c] = beta[c] - mean * g;
  }
}

// out = relu(pre*scale + shift) + out(identity)
__global__ void k_epilogue(const float* __restrict__ pre, const float* __restrict__ scale,
                           const float* __restrict__ shift, float* __restrict__ out,
                           int n4, int c4mask) {
  int i = blockIdx.x * blockDim.x + threadIdx.x;
  if (i < n4) {
    int c4 = i & c4mask;
    f4 v  = ((const f4*)pre)[i];
    f4 sc = ((const f4*)scale)[c4];
    f4 sh = ((const f4*)shift)[c4];
    f4 id = ((f4*)out)[i];
    f4 r;
#pragma unroll
    for (int j = 0; j < 4; ++j) r[j] = fmaxf(v[j] * sc[j] + sh[j], 0.0f) + id[j];
    ((f4*)out)[i] = r;
  }
}

// ---------------------------------------------------------------------------
// Launch
// ---------------------------------------------------------------------------
extern "C" void kernel_launch(void* const* d_in, const int* in_sizes, int n_in,
                              void* d_out, int out_size, void* d_ws, size_t ws_size,
                              hipStream_t stream) {
  const float* x      = (const float*)d_in[0];
  const float* cheb_w = (const float*)d_in[1];
  const float* cheb_b = (const float*)d_in[2];
  const float* lin_w  = (const float*)d_in[3];
  const float* lin_b  = (const float*)d_in[4];
  const float* gamma  = (const float*)d_in[5];
  const float* beta   = (const float*)d_in[6];
  const int*   eidx   = (const int*)d_in[7];

  const int C    = 512;
  const int Nn   = in_sizes[0] / C;     // 20000
  const int Ecnt = in_sizes[7] / 2;     // 320000
  const int* src = eidx;
  const int* dst = eidx + Ecnt;
  const size_t nc = (size_t)Nn * C;

  // workspace layout (floats); ~125 MB total
  float* ws       = (float*)d_ws;
  float* deg      = ws;                  // Nn
  float* bn_sum   = ws + Nn;             // 512
  float* bn_sumsq = bn_sum + C;          // 512
  float* bn_scale = bn_sumsq + C;        // 512
  float* bn_shift = bn_scale + C;        // 512
  float* norm     = bn_shift + C;        // Ecnt
  float* tx1      = norm + Ecnt;         // Nn*C
  float* tx2      = tx1 + nc;            // Nn*C
  float* pre      = tx2 + nc;            // Nn*C
  float* out      = (float*)d_out;

  // 1. zero deg + BN accumulators + tx1
  k_zero<<<(Nn + 4 * C + 255) / 256, 256, 0, stream>>>(deg, Nn + 4 * C);
  k_zero<<<((int)nc + 255) / 256, 256, 0, stream>>>(tx1, (int)nc);

  // 2. degree + edge weights
  k_deg<<<(Ecnt + 255) / 256, 256, 0, stream>>>(src, deg, Ecnt);
  k_norm<<<(Ecnt + 255) / 256, 256, 0, stream>>>(src, dst, deg, norm, Ecnt);

  // 3. tx1 = L_hat x ; tx2 = 2 L_hat tx1 - x
  k_spmm<<<Ecnt, 128, 0, stream>>>(src, dst, norm, x, tx1, 1.0f, C);
  k_negcopy<<<((int)(nc / 4) + 255) / 256, 256, 0, stream>>>(x, tx2, (int)(nc / 4));
  k_spmm<<<Ecnt, 128, 0, stream>>>(src, dst, norm, tx1, tx2, 2.0f, C);

  // 4. WMMA GEMMs: pre = x@W0 + tx1@W1 + tx2@W2 + cheb_b (with BN stats)
  //               out = x@lin_w + lin_b (identity path)
  dim3 gg((Nn + BM - 1) / BM, C / BN);
  k_gemm<<<gg, 256, 0, stream>>>(x, tx1, tx2,
                                 cheb_w, cheb_w + 262144, cheb_w + 524288,
                                 cheb_b, pre, Nn, C, C, 3, bn_sum, bn_sumsq, 1);
  k_gemm<<<gg, 256, 0, stream>>>(x, x, x, lin_w, lin_w, lin_w,
                                 lin_b, out, Nn, C, C, 1, bn_sum, bn_sumsq, 0);

  // 5. BN finalize + fused BN/ReLU/residual epilogue
  k_bnfin<<<(C + 255) / 256, 256, 0, stream>>>(bn_sum, bn_sumsq, gamma, beta,
                                               bn_scale, bn_shift, Nn, C, 1e-5f);
  k_epilogue<<<((int)(nc / 4) + 255) / 256, 256, 0, stream>>>(pre, bn_scale, bn_shift,
                                                              out, (int)(nc / 4), 127);
}
